// Agent_29068338659372
// MI455X (gfx1250) — matrix operand order repaired and checked by reading, not compiled, not run on hardware
//
#include <hip/hip_runtime.h>
#include <stdint.h>

// ---------------- WMMA / vector types ----------------
typedef __attribute__((ext_vector_type(16))) _Float16 v16h;
typedef __attribute__((ext_vector_type(8)))  _Float16 v8h;
typedef __attribute__((ext_vector_type(8)))  float    v8f;
typedef __attribute__((ext_vector_type(4))) unsigned int u32x4;
typedef __attribute__((ext_vector_type(8))) int          i32x8;
typedef __attribute__((ext_vector_type(4))) int          i32x4;

#define WMMA_F16(a,b,c) __builtin_amdgcn_wmma_f32_16x16x32_f16(false,(a),false,(b),(short)0,(c),false,false)

#if defined(__has_builtin)
#if __has_builtin(__builtin_amdgcn_tensor_load_to_lds) && __has_builtin(__builtin_amdgcn_s_wait_tensorcnt)
#define HAVE_TDM 1
#endif
#endif

// A-fragment (16x32 f16, M x K) per ISA layout:
// lane<16 : K chunks [0..7] and [16..23]; lane>=16 : [8..15] and [24..31]
static __device__ __forceinline__ v16h load_afrag(const _Float16* __restrict__ row, int kt, int hi){
  union { v16h v; v8h h[2]; } u;
  u.h[0] = *(const v8h*)(row + kt + 8*hi);
  u.h[1] = *(const v8h*)(row + kt + 16 + 8*hi);
  return u.v;
}
// same fragment but from an LDS tile laid out [m][32]
static __device__ __forceinline__ v16h lds_afrag(const _Float16* base, int m, int hi){
  union { v16h v; v8h h[2]; } u;
  u.h[0] = *(const v8h*)(base + m*32 + 8*hi);
  u.h[1] = *(const v8h*)(base + m*32 + 16 + 8*hi);
  return u.v;
}

// ---------------- prep kernels ----------------
__global__ void k_prep_w1(const float* __restrict__ w, _Float16* __restrict__ w1h){
  int i = blockIdx.x*256 + threadIdx.x;           // 65536
  int m = i >> 8, k = i & 255;
  float v = (k < 243) ? w[m*243 + k] : 0.f;
  w1h[i] = (_Float16)v;
}

__global__ void k_im2col(const float* __restrict__ x, _Float16* __restrict__ col){
  int i = blockIdx.x*256 + threadIdx.x;           // 26,214,400
  int k = i & 255; int n = i >> 8;
  int b = n / 400; int pos = n % 400; int y = pos/20, xx = pos%20;
  float v = 0.f;
  if (k < 243){
    int ci = k/81, rr = k%81, ky = rr/9, kx = rr%9;
    v = x[((b*3+ci)*28 + y+ky)*28 + xx+kx];
  }
  col[i] = (_Float16)v;
}

__global__ void k_prep_w2(const float* __restrict__ pw, _Float16* __restrict__ w2h){
  int i = blockIdx.x*256 + threadIdx.x;           // 5,308,416
  int m = i / 20736; int k = i % 20736;
  int ci = k & 255; int t = k >> 8; int ky = t/9, kx = t - 9*ky;
  w2h[i] = (_Float16)pw[((m*256 + ci)*9 + ky)*9 + kx];
}

// ---------------- GEMM1: conv1 (M=256,N=102400,K=256), bias+relu, out NHWC f16 ----------------
__global__ __launch_bounds__(256) void k_gemm1(
    const _Float16* __restrict__ A, const _Float16* __restrict__ Bm,
    const float* __restrict__ bias, _Float16* __restrict__ h16)
{
  __shared__ __attribute__((aligned(32))) _Float16 Bs[16*32];   // [n][k]
  const int tid  = threadIdx.x;
  const int lane = tid & 31, wv = tid >> 5;
  const int n0   = blockIdx.x << 4;
  const int e = tid << 1, nl = e >> 5, koff = e & 31;
  const _Float16* bp = Bm + (size_t)(n0 + nl)*256 + koff;
  const int r16 = lane & 15, hi = lane >> 4;
  const _Float16* pa0 = A + (size_t)(wv*32 + r16)*256;
  const _Float16* pa1 = pa0 + 16*256;
  v8f acc0 = {}, acc1 = {};
  for (int kt = 0; kt < 256; kt += 32){
    *(unsigned int*)&Bs[nl*32 + koff] = *(const unsigned int*)(bp + kt);
    __syncthreads();
    v16h bf = __builtin_shufflevector(*(const v8h*)&Bs[r16*32 + hi*8], *(const v8h*)&Bs[r16*32 + 16 + hi*8], 0,1,2,3,4,5,6,7,8,9,10,11,12,13,14,15);
    v16h a0 = load_afrag(pa0, kt, hi);
    v16h a1 = load_afrag(pa1, kt, hi);
    acc0 = WMMA_F16(a0, bf, acc0);
    acc1 = WMMA_F16(a1, bf, acc1);
    __syncthreads();
  }
  const int n  = n0 + r16;
  const int m0 = wv*32 + hi*8;
  v8h o0, o1;
  #pragma unroll
  for (int i = 0; i < 8; ++i){
    float z0 = acc0[i] + bias[m0+i];
    float z1 = acc1[i] + bias[m0+16+i];
    o0[i] = (_Float16)fmaxf(z0, 0.f);
    o1[i] = (_Float16)fmaxf(z1, 0.f);
  }
  *(v8h*)(h16 + (size_t)n*256 + m0)      = o0;
  *(v8h*)(h16 + (size_t)n*256 + m0 + 16) = o1;
}

// ---------------- GEMM2: primary caps implicit conv (M=256,N=9216,K=20736) ----------------
// Block tile 256(M) x 32(N); 8 waves, each 32x32 -> 4 WMMA accumulators.
// Double-buffered LDS; A tile (256x32 f16 = 16KB) DMA'd by the Tensor Data Mover.
__global__ __launch_bounds__(256) void k_gemm2(
    const _Float16* __restrict__ A, const _Float16* __restrict__ h16,
    const float* __restrict__ bias, float* __restrict__ out2)
{
  __shared__ __attribute__((aligned(64))) _Float16 Ab[2][256*32];  // [m][k]
  __shared__ __attribute__((aligned(64))) _Float16 Bb[2][32*32];   // [n][k]
  const int tid  = threadIdx.x;
  const int lane = tid & 31, wv = tid >> 5;
  const int r16  = lane & 15, hi = lane >> 4;
  const int n0   = blockIdx.x << 5;

  // cooperative B gather: each thread owns 4 halves of the 32x32 tile
  const int e = tid << 2, nl = e >> 5, koff = e & 31;
  const int ng = n0 + nl;
  const int b = ng/36, pos = ng%36, oy = pos/6, ox = pos%6;
  const _Float16* hb = h16 + ((size_t)b*400 + (size_t)(2*oy)*20 + (size_t)(2*ox))*256;

  v8f acc00 = {}, acc01 = {}, acc10 = {}, acc11 = {};

  auto stageB = [&](int kt, int buf){
    int kg = kt + koff;
    int ci = kg & 255; int t2 = kg >> 8; int ky = t2/9, kx = t2 - 9*ky;
    *(uint64_t*)&Bb[buf][nl*32 + koff] =
        *(const uint64_t*)(hb + (ky*20 + kx)*256 + ci);
  };

#if HAVE_TDM
  const unsigned ldsA0 = (unsigned)(size_t)&Ab[0][0];
  const unsigned ldsA1 = (unsigned)(size_t)&Ab[1][0];
  const unsigned long long gbase = (unsigned long long)(size_t)A;
  auto tdmLoadA = [&](int kt, int buf){
    // D# per CDNA5 ISA: group0 {count=1, lds_addr, global_addr, type=2}
    // group1 {data_size=2B, tensor_dim0=20736, tensor_dim1=256, tile 32x256, dim0_stride=20736}
    unsigned long long ga = gbase + (unsigned long long)(2*kt);
    u32x4 g0; i32x8 g1; i32x4 g2 = {0,0,0,0}, g3 = {0,0,0,0};
    i32x8 g4 = {0,0,0,0,0,0,0,0};
    g0[0] = 1u;                                            // count=1, normal (non-restore) mode
    g0[1] = buf ? ldsA1 : ldsA0;                           // lds_addr
    g0[2] = (unsigned)(ga & 0xffffffffu);                  // global_addr[31:0]
    g0[3] = (unsigned)((ga >> 32) & 0x1ffffffu) | (2u<<30);// global_addr[56:32] | type=2
    g1[0] = (int)(1u << 16);                               // data_size code 1 (2 bytes)
    g1[1] = (int)((20736u & 0xffffu) << 16);               // tensor_dim0[15:0]
    g1[2] = (int)((20736u >> 16) | (256u << 16));          // tensor_dim0[31:16] | tensor_dim1[15:0]
    g1[3] = (int)(32u << 16);                              // tensor_dim1[31:16]=0 | tile_dim0=32
    g1[4] = (int)256u;                                     // tile_dim1=256, tile_dim2=0
    g1[5] = (int)20736u;                                   // tensor_dim0_stride[31:0]
    g1[6] = 0;                                             // stride hi bits / dim1_stride lo
    g1[7] = 0;
    __builtin_amdgcn_tensor_load_to_lds(g0, g1, g2, g3, g4, 0);
  };
#endif

  const int NSTEP = 20736/32;   // 648
  // prologue: stage step 0
  stageB(0, 0);
#if HAVE_TDM
  if (wv == 0) tdmLoadA(0, 0);
#endif

  for (int i = 0; i < NSTEP; ++i){
    const int kt  = i << 5;
    const int buf = i & 1;
#if HAVE_TDM
    if (wv == 0) __builtin_amdgcn_s_wait_tensorcnt(0);   // tile i landed in LDS
#endif
    __syncthreads();                                      // one barrier per K-step
    if (i + 1 < NSTEP){                                   // stage i+1 into back buffer
      stageB(kt + 32, buf ^ 1);
#if HAVE_TDM
      if (wv == 0) tdmLoadA(kt + 32, buf ^ 1);
#endif
    }
#if HAVE_TDM
    const _Float16* Arow = &Ab[buf][0];
    v16h a0 = lds_afrag(Arow, wv*32 + r16,      hi);
    v16h a1 = lds_afrag(Arow, wv*32 + 16 + r16, hi);
#else
    __builtin_prefetch(A + (size_t)(wv*32 + r16)*20736 + kt + 256, 0, 1);
    v16h a0 = load_afrag(A + (size_t)(wv*32 + r16)*20736,      kt, hi);
    v16h a1 = load_afrag(A + (size_t)(wv*32 + 16 + r16)*20736, kt, hi);
#endif
    v16h b0 = __builtin_shufflevector(*(const v8h*)&Bb[buf][r16*32 + hi*8], *(const v8h*)&Bb[buf][r16*32 + 16 + hi*8], 0,1,2,3,4,5,6,7,8,9,10,11,12,13,14,15);
    v16h b1 = __builtin_shufflevector(*(const v8h*)&Bb[buf][(16 + r16)*32 + hi*8], *(const v8h*)&Bb[buf][(16 + r16)*32 + 16 + hi*8], 0,1,2,3,4,5,6,7,8,9,10,11,12,13,14,15);
    acc00 = WMMA_F16(a0, b0, acc00);
    acc01 = WMMA_F16(a0, b1, acc01);
    acc10 = WMMA_F16(a1, b0, acc10);
    acc11 = WMMA_F16(a1, b1, acc11);
  }

  const int m0 = wv*32 + hi*8;
  float* p0 = out2 + (size_t)(n0 + r16)*256;
  float* p1 = out2 + (size_t)(n0 + 16 + r16)*256;
  #pragma unroll
  for (int i2 = 0; i2 < 8; ++i2){
    p0[m0+i2]    = acc00[i2] + bias[m0+i2];
    p1[m0+i2]    = acc01[i2] + bias[m0+i2];
    p0[m0+16+i2] = acc10[i2] + bias[m0+16+i2];
    p1[m0+16+i2] = acc11[i2] + bias[m0+16+i2];
  }
}

// ---------------- squash/reshape: out2[n=b*36+pos][ch] -> u[b][r][p] ----------------
__global__ void k_build_u(const float* __restrict__ out2, float* __restrict__ u){
  int id = blockIdx.x*256 + threadIdx.x;          // 294912
  int b = id / 1152, r = id % 1152;
  float vv[8]; float sq = 0.f;
  #pragma unroll
  for (int p = 0; p < 8; ++p){
    int flat = r*8 + p; int ch = flat/36, pos = flat%36;
    float t = out2[(size_t)(b*36 + pos)*256 + ch];
    vv[p] = t; sq += t*t;
  }
  float f = sqrtf(sq)/(1.f + sq);
  #pragma unroll
  for (int p = 0; p < 8; ++p) u[((size_t)b*1152 + r)*8 + p] = vv[p]*f;
}

__global__ void k_zero(float* __restrict__ p){
  int i = blockIdx.x*256 + threadIdx.x;           // 11520
  p[i] = 0.f;
}

// ---------------- routing ----------------
__global__ void k_softmax(const float* __restrict__ bij, float* __restrict__ c){
  const int k = blockIdx.x, t = threadIdx.x;
  __shared__ float red[256];
  float mx = -1e30f;
  for (int r = t; r < 1152; r += 256) mx = fmaxf(mx, bij[r*10 + k]);
  red[t] = mx; __syncthreads();
  for (int s = 128; s > 0; s >>= 1){ if (t < s) red[t] = fmaxf(red[t], red[t+s]); __syncthreads(); }
  mx = red[0]; __syncthreads();
  float sm = 0.f;
  for (int r = t; r < 1152; r += 256) sm += expf(bij[r*10 + k] - mx);
  red[t] = sm; __syncthreads();
  for (int s = 128; s > 0; s >>= 1){ if (t < s) red[t] += red[t+s]; __syncthreads(); }
  sm = red[0]; __syncthreads();
  float inv = 1.f/sm;
  for (int r = t; r < 1152; r += 256) c[r*10 + k] = expf(bij[r*10 + k] - mx)*inv;
}

__global__ __launch_bounds__(160) void k_suv(
    const float* __restrict__ W, const float* __restrict__ u,
    const float* __restrict__ c, float* __restrict__ vout, float* __restrict__ vbuf)
{
  const int b = blockIdx.x, t = threadIdx.x;
  const int k = t >> 4, o = t & 15;
  const float* ub = u + (size_t)b*1152*8;
  float s = 0.f;
  for (int r = 0; r < 1152; ++r){
    const float* wr = W + ((size_t)(r*10 + k)*16 + o)*8;
    const float* ur = ub + r*8;
    float uh = 0.f;
    #pragma unroll
    for (int i = 0; i < 8; ++i) uh += wr[i]*ur[i];
    s += c[r*10 + k]*uh;
  }
  float v = s*fabsf(s)/(1.f + s*s);
  vout[(b*10 + k)*16 + o] = v;
  vbuf[(b*10 + k)*16 + o] = v;
}

__global__ __launch_bounds__(256) void k_bupdate(
    const float* __restrict__ W, const float* __restrict__ u,
    const float* __restrict__ vbuf, float* __restrict__ bij)
{
  const int r = blockIdx.x, t = threadIdx.x;
  __shared__ float Ws[1280];
  __shared__ float red[256];
  for (int i = t; i < 1280; i += 256) Ws[i] = W[(size_t)r*1280 + i];
  __syncthreads();
  float ur[8];
  #pragma unroll
  for (int i = 0; i < 8; ++i) ur[i] = u[((size_t)t*1152 + r)*8 + i];
  for (int k = 0; k < 10; ++k){
    float acc = 0.f;
    for (int o = 0; o < 16; ++o){
      const float* wr = &Ws[(k*16 + o)*8];
      float uh = 0.f;
      #pragma unroll
      for (int i = 0; i < 8; ++i) uh += wr[i]*ur[i];
      acc += uh * vbuf[(t*10 + k)*16 + o];
    }
    red[t] = acc; __syncthreads();
    for (int s = 128; s > 0; s >>= 1){ if (t < s) red[t] += red[t+s]; __syncthreads(); }
    if (t == 0) bij[r*10 + k] += red[0]*(1.f/256.f);
    __syncthreads();
  }
}

// ---------------- host ----------------
extern "C" void kernel_launch(void* const* d_in, const int* in_sizes, int n_in,
                              void* d_out, int out_size, void* d_ws, size_t ws_size,
                              hipStream_t stream)
{
  (void)in_sizes; (void)n_in; (void)out_size; (void)ws_size;
  const float* x  = (const float*)d_in[0];   // [256,3,28,28]
  const float* w1 = (const float*)d_in[1];   // [256,3,9,9]
  const float* b1 = (const float*)d_in[2];   // [256]
  const float* pw = (const float*)d_in[3];   // [8,32,256,9,9] == [256,256,9,9]
  const float* pb = (const float*)d_in[4];   // [8,32] == [256]
  const float* Wr = (const float*)d_in[5];   // [1152,10,16,8]
  float* out = (float*)d_out;                // [256,10,16]

  char* ws = (char*)d_ws;
  size_t off = 0;
  auto alloc = [&](size_t bytes)->char*{
    char* p = ws + off; off = (off + bytes + 255) & ~(size_t)255; return p;
  };
  _Float16* w1h  = (_Float16*)alloc((size_t)256*256*2);
  _Float16* col1 = (_Float16*)alloc((size_t)102400*256*2);
  _Float16* h16  = (_Float16*)alloc((size_t)102400*256*2);
  _Float16* w2h  = (_Float16*)alloc((size_t)256*20736*2);
  float*    out2 = (float*)alloc((size_t)9216*256*4);
  float*    u    = (float*)alloc((size_t)256*1152*8*4);
  float*    bij  = (float*)alloc((size_t)1152*10*4);
  float*    cc   = (float*)alloc((size_t)1152*10*4);
  float*    vbuf = (float*)alloc((size_t)256*160*4);

  k_prep_w1<<<256,    256, 0, stream>>>(w1, w1h);
  k_im2col <<<102400, 256, 0, stream>>>(x, col1);
  k_prep_w2<<<20736,  256, 0, stream>>>(pw, w2h);
  k_gemm1  <<<6400,   256, 0, stream>>>(w1h, col1, b1, h16);
  k_gemm2  <<<288,    256, 0, stream>>>(w2h, h16, pb, out2);
  k_build_u<<<1152,   256, 0, stream>>>(out2, u);
  k_zero   <<<45,     256, 0, stream>>>(bij);
  for (int it = 0; it < 3; ++it){
    k_softmax<<<10,   256, 0, stream>>>(bij, cc);
    k_suv    <<<256,  160, 0, stream>>>(Wr, u, cc, out, vbuf);
    if (it < 2) k_bupdate<<<1152, 256, 0, stream>>>(Wr, u, vbuf, bij);
  }
}
